// LSTMTagger_2216203125241
// MI455X (gfx1250) — compile-verified, matrix operand order checked by reading, not run on hardware
//
#include <hip/hip_runtime.h>
#include <hip/hip_bf16.h>
#include <math.h>

// Problem constants (from reference)
#define S_LEN 2048
#define L_CH  16
#define DW_   512
#define DC_   128
#define H_    512
#define TAGS  64
#define TC_   (S_LEN * L_CH)   // 32768 flattened char steps
#define CPAD  132              // padded LDS row stride (bf16 elems): 264B = 66 DWORDs
                               // -> lane j hits bank (2j mod 64): conflict-free, 4B aligned

typedef __attribute__((ext_vector_type(16))) __bf16 v16bf;
typedef __attribute__((ext_vector_type(8)))  float  v8f;

__device__ __forceinline__ float2 bf16pair_to_f2(unsigned int u) {
    union { unsigned int u; __bf16 h[2]; } cv;
    cv.u = u;
    return make_float2((float)cv.h[0], (float)cv.h[1]);
}

// ---------------------------------------------------------------------------
// Build a 16-element bf16 WMMA fragment for one lane from a row-major row `p`.
// Per CDNA5 ISA 16-bit A/B layout:
//   lanes 0-15  (hi=0): elements 0..7 = K k0+0..7,  8..15 = K k0+16..23
//   lanes 16-31 (hi=1): elements 0..7 = K k0+8..15, 8..15 = K k0+24..31
// ---------------------------------------------------------------------------
__device__ __forceinline__ v16bf frag16(const float* __restrict__ p, int k0, int hi) {
    const float4* q0 = reinterpret_cast<const float4*>(p + k0 + hi * 8);
    const float4* q1 = reinterpret_cast<const float4*>(p + k0 + 16 + hi * 8);
    float4 a0 = q0[0], a1 = q0[1];
    float4 b0 = q1[0], b1 = q1[1];
    v16bf r;
    r[0]  = (__bf16)a0.x; r[1]  = (__bf16)a0.y; r[2]  = (__bf16)a0.z; r[3]  = (__bf16)a0.w;
    r[4]  = (__bf16)a1.x; r[5]  = (__bf16)a1.y; r[6]  = (__bf16)a1.z; r[7]  = (__bf16)a1.w;
    r[8]  = (__bf16)b0.x; r[9]  = (__bf16)b0.y; r[10] = (__bf16)b0.z; r[11] = (__bf16)b0.w;
    r[12] = (__bf16)b1.x; r[13] = (__bf16)b1.y; r[14] = (__bf16)b1.z; r[15] = (__bf16)b1.w;
    return r;
}

// ---------------------------------------------------------------------------
// Kernel 1: XGc[32768,512] = gather(char_emb, chars) @ Wih_c^T + b_c
// One wave per 16x16 output tile; 4 waves per block. K = 128 (4 WMMA steps).
// ---------------------------------------------------------------------------
__global__ void k_char_xg(const int* __restrict__ chars,
                          const float* __restrict__ char_emb,
                          const float* __restrict__ Wih,
                          const float* __restrict__ bias,
                          float* __restrict__ XG) {
    const int lane = threadIdx.x & 31;
    const int wave = threadIdx.x >> 5;
    const int tile = blockIdx.x * 4 + wave;   // 65536 tiles total
    const int mt = tile >> 5;                 // 2048 M-tiles
    const int nt = tile & 31;                 // 32 N-tiles
    const int hi = lane >> 4;
    const int lo = lane & 15;
    const int m = mt * 16 + lo;
    const int n = nt * 16 + lo;

    const float* arow = char_emb + (size_t)chars[m] * DC_;
    const float* brow = Wih + (size_t)n * DC_;

    v8f acc = {};
#pragma unroll
    for (int kk = 0; kk < DC_; kk += 32) {
        v16bf af = frag16(arow, kk, hi);
        v16bf bf = frag16(brow, kk, hi);
        acc = __builtin_amdgcn_wmma_f32_16x16x32_bf16(false, af, false, bf,
                                                      (short)0, acc, false, false);
    }
    const float bv = bias[n];
#pragma unroll
    for (int r = 0; r < 8; ++r) {
        const int row = mt * 16 + hi * 8 + r;   // C layout: VGPR r -> M = r + 8*hi
        XG[(size_t)row * (4 * DC_) + n] = acc[r] + bv;
    }
}

// ---------------------------------------------------------------------------
// Kernel 2: char LSTM recurrence, 32768 sequential steps, one workgroup.
// Whh_c staged in LDS as bf16 with padded row stride (conflict-free banks);
// inner loop consumes weight pairs as single aligned b32 LDS loads.
// Next step's xg row is software-pipelined into registers under the dot.
// ---------------------------------------------------------------------------
__global__ void k_char_lstm(const float* __restrict__ XG,
                            const float* __restrict__ Whh,
                            float* __restrict__ char_rep) {
    extern __shared__ char smem[];
    __bf16* sW = (__bf16*)smem;                            // [512][CPAD]
    float*  hv = (float*)(smem + (size_t)512 * CPAD * 2);  // [128]
    float*  gv = hv + DC_;                                 // [512]

    const int j = threadIdx.x;   // 0..511, owns gate row j
    for (int k = 0; k < DC_; ++k) sW[j * CPAD + k] = (__bf16)Whh[j * DC_ + k];
    if (j < DC_) hv[j] = 0.0f;
    float c = 0.0f;
    __syncthreads();

    const __bf16* wrow = sW + (size_t)j * CPAD;
    float xg_cur = XG[j];
    for (int t = 0; t < TC_; ++t) {
        float acc = xg_cur;
        // issue next step's load early; latency hides under the dot product
        float xg_next = (t + 1 < TC_) ? XG[(size_t)(t + 1) * 512 + j] : 0.0f;
#pragma unroll 8
        for (int k = 0; k < DC_; k += 2) {
            const float2 w = bf16pair_to_f2(
                *reinterpret_cast<const unsigned int*>(wrow + k));
            acc += w.x * hv[k] + w.y * hv[k + 1];
        }
        gv[j] = acc;
        __syncthreads();
        if (j < DC_) {
            const float ig = 1.0f / (1.0f + __expf(-gv[j]));
            const float fg = 1.0f / (1.0f + __expf(-gv[DC_ + j]));
            const float gg = tanhf(gv[2 * DC_ + j]);
            const float og = 1.0f / (1.0f + __expf(-gv[3 * DC_ + j]));
            c = fg * c + ig * gg;
            const float h = og * tanhf(c);
            hv[j] = h;
            if ((t & (L_CH - 1)) == (L_CH - 1))          // last char of each word
                char_rep[(size_t)(t >> 4) * DC_ + j] = h;
        }
        __syncthreads();
        xg_cur = xg_next;
    }
}

// ---------------------------------------------------------------------------
// Kernel 3: XGw[2048,2048] = [we(sentence) ‖ char_rep] @ Wih_w^T + b_w
// ---------------------------------------------------------------------------
__global__ void k_word_xg(const int* __restrict__ sentence,
                          const float* __restrict__ word_emb,
                          const float* __restrict__ char_rep,
                          const float* __restrict__ Wih,
                          const float* __restrict__ bias,
                          float* __restrict__ XG) {
    const int lane = threadIdx.x & 31;
    const int wave = threadIdx.x >> 5;
    const int tile = blockIdx.x * 4 + wave;   // 16384 tiles
    const int mt = tile >> 7;                 // 128 M-tiles
    const int nt = tile & 127;                // 128 N-tiles
    const int hi = lane >> 4;
    const int lo = lane & 15;
    const int m = mt * 16 + lo;
    const int n = nt * 16 + lo;

    const float* we   = word_emb + (size_t)sentence[m] * DW_;
    const float* cr   = char_rep + (size_t)m * DC_;
    const float* brow = Wih + (size_t)n * (DW_ + DC_);

    v8f acc = {};
    for (int kk = 0; kk < DW_; kk += 32) {
        v16bf af = frag16(we, kk, hi);
        v16bf bf = frag16(brow, kk, hi);
        acc = __builtin_amdgcn_wmma_f32_16x16x32_bf16(false, af, false, bf,
                                                      (short)0, acc, false, false);
    }
#pragma unroll
    for (int kk = 0; kk < DC_; kk += 32) {
        v16bf af = frag16(cr, kk, hi);
        v16bf bf = frag16(brow, DW_ + kk, hi);
        acc = __builtin_amdgcn_wmma_f32_16x16x32_bf16(false, af, false, bf,
                                                      (short)0, acc, false, false);
    }
    const float bv = bias[n];
#pragma unroll
    for (int r = 0; r < 8; ++r) {
        const int row = mt * 16 + hi * 8 + r;
        XG[(size_t)row * (4 * H_) + n] = acc[r] + bv;
    }
}

// ---------------------------------------------------------------------------
// Kernel 4a: one-time transpose + bf16 convert of Whh_w [2048,512] (f32)
// into WT [512][2048] (bf16, 2MB, L2-resident) for coalesced recurrence reads.
// ---------------------------------------------------------------------------
__global__ void k_transpose_whw(const float* __restrict__ W, __bf16* __restrict__ WT) {
    const int idx = blockIdx.x * 256 + threadIdx.x;   // 0 .. 2048*512-1
    const int row = idx >> 9;     // gate row j: 0..2047
    const int k   = idx & 511;    // 0..511
    WT[(size_t)k * (4 * H_) + row] = (__bf16)W[idx];
}

// ---------------------------------------------------------------------------
// Kernel 4b: word LSTM recurrence, 2048 sequential steps, one workgroup of
// 1024 threads. Thread j owns ADJACENT gate rows 2j, 2j+1 so one coalesced
// b32 load of transposed bf16 weights yields both per-k weights
// (consecutive lanes -> consecutive DWORDs). h[k] is an LDS broadcast.
// ---------------------------------------------------------------------------
__global__ void k_word_lstm(const float* __restrict__ XG,
                            const __bf16* __restrict__ WT,
                            float* __restrict__ wh) {
    __shared__ float hv[H_];
    __shared__ float gv[4 * H_];
    const int j = threadIdx.x;   // 0..1023, owns gate rows 2j and 2j+1
    if (j < H_) hv[j] = 0.0f;
    float c = 0.0f;
    __syncthreads();

    float2 xg_cur = *reinterpret_cast<const float2*>(XG + 2 * j);
    for (int t = 0; t < S_LEN; ++t) {
        float a0 = xg_cur.x, a1 = xg_cur.y;
        const float2 xg_next = (t + 1 < S_LEN)
            ? *reinterpret_cast<const float2*>(XG + (size_t)(t + 1) * (4 * H_) + 2 * j)
            : make_float2(0.0f, 0.0f);
#pragma unroll 4
        for (int k = 0; k < H_; ++k) {
            const float2 w = bf16pair_to_f2(*reinterpret_cast<const unsigned int*>(
                WT + (size_t)k * (4 * H_) + 2 * j));
            const float hk = hv[k];
            a0 += w.x * hk;
            a1 += w.y * hk;
        }
        *reinterpret_cast<float2*>(gv + 2 * j) = make_float2(a0, a1);
        __syncthreads();
        if (j < H_) {
            const float ig = 1.0f / (1.0f + __expf(-gv[j]));
            const float fg = 1.0f / (1.0f + __expf(-gv[H_ + j]));
            const float gg = tanhf(gv[2 * H_ + j]);
            const float og = 1.0f / (1.0f + __expf(-gv[3 * H_ + j]));
            c = fg * c + ig * gg;
            const float h = og * tanhf(c);
            hv[j] = h;
            wh[(size_t)t * H_ + j] = h;
        }
        __syncthreads();
        xg_cur = xg_next;
    }
}

// ---------------------------------------------------------------------------
// Kernel 5: tag_space[2048,64] = wh @ W_tag^T + b_tag (WMMA, K=512)
// ---------------------------------------------------------------------------
__global__ void k_tag_gemm(const float* __restrict__ wh,
                           const float* __restrict__ Wtag,
                           const float* __restrict__ btag,
                           float* __restrict__ ts) {
    const int lane = threadIdx.x & 31;
    const int wave = threadIdx.x >> 5;
    const int tile = blockIdx.x * 4 + wave;   // 512 tiles
    const int mt = tile >> 2;                 // 128 M-tiles
    const int nt = tile & 3;                  // 4 N-tiles
    const int hi = lane >> 4;
    const int lo = lane & 15;
    const int m = mt * 16 + lo;
    const int n = nt * 16 + lo;

    const float* arow = wh + (size_t)m * H_;
    const float* brow = Wtag + (size_t)n * H_;

    v8f acc = {};
    for (int kk = 0; kk < H_; kk += 32) {
        v16bf af = frag16(arow, kk, hi);
        v16bf bf = frag16(brow, kk, hi);
        acc = __builtin_amdgcn_wmma_f32_16x16x32_bf16(false, af, false, bf,
                                                      (short)0, acc, false, false);
    }
    const float bv = btag[n];
#pragma unroll
    for (int r = 0; r < 8; ++r) {
        const int row = mt * 16 + hi * 8 + r;
        ts[(size_t)row * TAGS + n] = acc[r] + bv;
    }
}

// ---------------------------------------------------------------------------
// Kernel 6: rowwise log_softmax over 64 tags; one wave32 per row (2 cols/lane).
// ---------------------------------------------------------------------------
__global__ void k_logsoftmax(const float* __restrict__ ts, float* __restrict__ out) {
    const int lane = threadIdx.x & 31;
    const int row = blockIdx.x * 8 + (threadIdx.x >> 5);
    const float* x = ts + (size_t)row * TAGS;
    float x0 = x[lane], x1 = x[lane + 32];
    float mx = fmaxf(x0, x1);
#pragma unroll
    for (int off = 16; off > 0; off >>= 1) mx = fmaxf(mx, __shfl_xor(mx, off, 32));
    float s = __expf(x0 - mx) + __expf(x1 - mx);
#pragma unroll
    for (int off = 16; off > 0; off >>= 1) s += __shfl_xor(s, off, 32);
    const float lse = mx + __logf(s);
    out[(size_t)row * TAGS + lane] = x0 - lse;
    out[(size_t)row * TAGS + lane + 32] = x1 - lse;
}

// ---------------------------------------------------------------------------
extern "C" void kernel_launch(void* const* d_in, const int* in_sizes, int n_in,
                              void* d_out, int out_size, void* d_ws, size_t ws_size,
                              hipStream_t stream) {
    const int*   sentence = (const int*)d_in[0];
    const int*   chars    = (const int*)d_in[1];
    const float* word_emb = (const float*)d_in[2];
    const float* char_emb = (const float*)d_in[3];
    const float* Wih_c    = (const float*)d_in[4];
    const float* Whh_c    = (const float*)d_in[5];
    const float* b_c      = (const float*)d_in[6];
    const float* Wih_w    = (const float*)d_in[7];
    const float* Whh_w    = (const float*)d_in[8];
    const float* b_w      = (const float*)d_in[9];
    const float* W_tag    = (const float*)d_in[10];
    const float* b_tag    = (const float*)d_in[11];
    float* out = (float*)d_out;

    float* ws       = (float*)d_ws;
    float* XGc      = ws;                                   // 32768*512  (64 MB)
    float* char_rep = XGc + (size_t)TC_ * (4 * DC_);        // 2048*128   (1 MB)
    float* XGw      = char_rep + (size_t)S_LEN * DC_;       // 2048*2048  (16 MB)
    float* whbuf    = XGw + (size_t)S_LEN * (4 * H_);       // 2048*512   (4 MB)
    float* tagsp    = whbuf + (size_t)S_LEN * H_;           // 2048*64    (0.5 MB)
    __bf16* WTw     = (__bf16*)(tagsp + (size_t)S_LEN * TAGS); // 512*2048 bf16 (2 MB)

    // 1) char-LSTM input GEMM (WMMA bf16): 65536 16x16 tiles, 4 waves/block
    k_char_xg<<<16384, 128, 0, stream>>>(chars, char_emb, Wih_c, b_c, XGc);

    // 2) char-LSTM recurrence (padded bf16 weights in dynamic LDS, ~134 KB)
    const size_t lds_bytes = (size_t)512 * CPAD * 2 + (size_t)(DC_ + 4 * DC_) * sizeof(float);
    k_char_lstm<<<1, 512, lds_bytes, stream>>>(XGc, Whh_c, char_rep);

    // 3) word-LSTM input GEMM with fused embedding gather + concat
    k_word_xg<<<4096, 128, 0, stream>>>(sentence, word_emb, char_rep, Wih_w, b_w, XGw);

    // 4a) transpose+convert Whh_w -> bf16 [512][2048] for coalesced reads
    k_transpose_whw<<<(2048 * 512) / 256, 256, 0, stream>>>(Whh_w, WTw);

    // 4b) word-LSTM recurrence
    k_word_lstm<<<1, 1024, 0, stream>>>(XGw, WTw, whbuf);

    // 5) tag projection GEMM (WMMA bf16)
    k_tag_gemm<<<128, 128, 0, stream>>>(whbuf, W_tag, b_tag, tagsp);

    // 6) log-softmax, one wave per row
    k_logsoftmax<<<256, 256, 0, stream>>>(tagsp, out);
}
